// ForcePai_532575945327
// MI455X (gfx1250) — matrix operand order repaired and checked by reading, not compiled, not run on hardware
//
#include <hip/hip_runtime.h>
#include <hip/hip_bf16.h>
#include <math.h>

#define N_ATOMS 20000
#define N_EDGES 320000
#define FDIM    128
#define F3      (3*FDIM)
#define NRBF    20
#define NLAY    3
#define PI_F    3.14159265358979f
#define CUT_F   5.0f
#define CDIV(a,b) (((a)+(b)-1)/(b))

typedef __bf16 bf16_t;
typedef __bf16 v16bf __attribute__((ext_vector_type(16)));
typedef __bf16 v8bf  __attribute__((ext_vector_type(8)));
typedef float  v8f   __attribute__((ext_vector_type(8)));

__device__ __forceinline__ bf16_t f2bf(float x) {
    unsigned u = __builtin_bit_cast(unsigned, x);
    unsigned r = u + 0x7fffu + ((u >> 16) & 1u);   // round-to-nearest-even
    unsigned short h = (unsigned short)(r >> 16);
    return __builtin_bit_cast(bf16_t, h);
}
__device__ __forceinline__ float bf2f(bf16_t b) {
    unsigned short h = __builtin_bit_cast(unsigned short, b);
    unsigned u = ((unsigned)h) << 16;
    return __builtin_bit_cast(float, u);
}
__device__ __forceinline__ float swishf(float x) { return x / (1.0f + expf(-x)); }

// ---------------------------------------------------------------------------
// bf16 WMMA GEMM, register-blocked: each wave computes a 32x64 output tile
// (2 M-subtiles x 4 N-subtiles = 8 accumulators). Per k-step: 2 A fragments +
// 4 B fragments feed 8 v_wmma_f32_16x16x32_bf16 -> 1.5 B128 loads per WMMA.
// W supplied pre-transposed as WT[Nc x K] (bf16) so all fragment loads are
// contiguous 16B. Requires M % 32 == 0, Nc % 64 == 0 (true for all uses).
// ---------------------------------------------------------------------------
template<int K, bool SWISH, bool OUT_BF16, bool ACCUM>
__global__ void wmma_gemm(const bf16_t* __restrict__ A, int lda,
                          const bf16_t* __restrict__ WT,
                          const float* __restrict__ bias,
                          float* __restrict__ Cf, bf16_t* __restrict__ Cb,
                          int ldc, int M, int Ncols)
{
    const int gwave = (int)((blockIdx.x * blockDim.x + threadIdx.x) >> 5);
    const int lane  = (int)(threadIdx.x & 31);
    const int tilesN = Ncols >> 6;                 // 64-wide N tiles
    const int totalTiles = (M >> 5) * tilesN;      // 32-tall M tiles
    if (gwave >= totalTiles) return;               // uniform per wave -> EXEC all-1s
    const int tM = gwave / tilesN;
    const int tN = gwave - tM * tilesN;
    const int half = lane >> 4;
    const int mr   = lane & 15;

    v8f c[2][4];
    #pragma unroll
    for (int j = 0; j < 4; j++) {
        const float bv = bias ? bias[tN * 64 + j * 16 + mr] : 0.0f;
        #pragma unroll
        for (int i = 0; i < 2; i++)
            #pragma unroll
            for (int r = 0; r < 8; r++) c[i][j][r] = bv;
    }

    const bf16_t* __restrict__ Arow0 = A + (size_t)(tM * 32 + mr) * lda;
    const bf16_t* __restrict__ Arow1 = Arow0 + (size_t)16 * lda;
    const bf16_t* __restrict__ Wrow[4];
    #pragma unroll
    for (int j = 0; j < 4; j++)
        Wrow[j] = WT + (size_t)(tN * 64 + j * 16 + mr) * K;

    #pragma unroll
    for (int k0 = 0; k0 < K; k0 += 32) {
        // A 16x32 bf16 layout: lane half h holds K = k0+8h..+7 and k0+16+8h..+7
        v16bf av[2];
        {
            v8bf a00 = *(const v8bf*)(Arow0 + k0 + half * 8);
            v8bf a01 = *(const v8bf*)(Arow0 + k0 + 16 + half * 8);
            v8bf a10 = *(const v8bf*)(Arow1 + k0 + half * 8);
            v8bf a11 = *(const v8bf*)(Arow1 + k0 + 16 + half * 8);
            #pragma unroll
            for (int t = 0; t < 8; t++) {
                av[0][t] = a00[t]; av[0][8 + t] = a01[t];
                av[1][t] = a10[t]; av[1][8 + t] = a11[t];
            }
        }
        // B 32x16 bf16 layout: lane half h holds K = k0+8h..+7 and k0+16+8h..+7
        v16bf bw[4];
        #pragma unroll
        for (int j = 0; j < 4; j++) {
            v8bf b0 = *(const v8bf*)(Wrow[j] + k0 + half * 8);
            v8bf b1 = *(const v8bf*)(Wrow[j] + k0 + 16 + half * 8);
            #pragma unroll
            for (int t = 0; t < 8; t++) { bw[j][t] = b0[t]; bw[j][8 + t] = b1[t]; }
        }
        #pragma unroll
        for (int i = 0; i < 2; i++)
            #pragma unroll
            for (int j = 0; j < 4; j++)
                c[i][j] = __builtin_amdgcn_wmma_f32_16x16x32_bf16(
                    false, av[i], false, bw[j], (short)0, c[i][j], false, false);
    }

    #pragma unroll
    for (int i = 0; i < 2; i++) {
        #pragma unroll
        for (int j = 0; j < 4; j++) {
            const int col = tN * 64 + j * 16 + mr;
            #pragma unroll
            for (int r = 0; r < 8; r++) {
                const int m = tM * 32 + i * 16 + half * 8 + r;  // C: VGPR r -> M = r+8*half
                const size_t off = (size_t)m * ldc + col;
                float x = c[i][j][r];
                if (ACCUM) x += Cf[off];
                if (SWISH) x = swishf(x);
                if (OUT_BF16) Cb[off] = f2bf(x); else Cf[off] = x;
            }
        }
    }
}

// ---------------------------------------------------------------------------
// Weight transpose + bf16 convert: W[K x Nc] (f32) -> WT[Nc x K] (bf16)
// ---------------------------------------------------------------------------
__global__ void w_transpose(const float* __restrict__ W, bf16_t* __restrict__ WT,
                            int K, int Nc)
{
    int idx = blockIdx.x * blockDim.x + threadIdx.x;
    if (idx >= K * Nc) return;
    int k = idx / Nc, n = idx - k * Nc;
    WT[(size_t)n * K + k] = f2bf(W[idx]);
}

__global__ void f32_to_bf16(const float* __restrict__ src, bf16_t* __restrict__ dst, int cnt)
{
    int idx = blockIdx.x * blockDim.x + threadIdx.x;
    if (idx < cnt) dst[idx] = f2bf(src[idx]);
}

// ---------------------------------------------------------------------------
// Edge geometry: r_ij = xyz[dst]-xyz[src], dist, unit, rbf, cos envelope
// ---------------------------------------------------------------------------
__global__ void edge_geom(const float* __restrict__ xyz, const int* __restrict__ nbrs,
                          float* __restrict__ unitb, float* __restrict__ rbfb,
                          float* __restrict__ envb)
{
    int e = blockIdx.x * blockDim.x + threadIdx.x;
    if (e >= N_EDGES) return;
    int src = nbrs[2 * e], dst = nbrs[2 * e + 1];
    float dx = xyz[dst * 3 + 0] - xyz[src * 3 + 0];
    float dy = xyz[dst * 3 + 1] - xyz[src * 3 + 1];
    float dz = xyz[dst * 3 + 2] - xyz[src * 3 + 2];
    float d = sqrtf(dx * dx + dy * dy + dz * dz + 3.0f * 1e-15f);
    float inv = 1.0f / d;
    unitb[e * 3 + 0] = dx * inv; unitb[e * 3 + 1] = dy * inv; unitb[e * 3 + 2] = dz * inv;
    #pragma unroll
    for (int j = 0; j < NRBF; j++)
        rbfb[(size_t)e * NRBF + j] = sinf((float)(j + 1) * (PI_F / CUT_F) * d) * inv;
    envb[e] = (d < CUT_F) ? 0.5f * (cosf(PI_F * d / CUT_F) + 1.0f) : 0.0f;
}

// e_ij = (rbf @ de_w + de_b) * env   (K=20 -> VALU)
__global__ void e_init(const float* __restrict__ rbfb, const float* __restrict__ envb,
                       const float* __restrict__ de_w, const float* __restrict__ de_b,
                       float* __restrict__ eij)
{
    int e = blockIdx.x, f = threadIdx.x;
    __shared__ float sr[NRBF];
    __shared__ float se;
    if (f < NRBF) sr[f] = rbfb[(size_t)e * NRBF + f];
    else if (f == NRBF) se = envb[e];
    __syncthreads();
    float acc = de_b[f];
    #pragma unroll
    for (int j = 0; j < NRBF; j++) acc += sr[j] * de_w[j * FDIM + f];
    eij[(size_t)e * FDIM + f] = acc * se;
}

// s = embed[z], v = 0, accumulators = 0
__global__ void init_nodes(const float* __restrict__ embed, const int* __restrict__ z,
                           float* __restrict__ s, float* __restrict__ v,
                           float* __restrict__ ds, float* __restrict__ dv)
{
    int idx = blockIdx.x * blockDim.x + threadIdx.x;
    if (idx >= N_ATOMS * F3) return;
    v[idx] = 0.0f; dv[idx] = 0.0f;
    if (idx < N_ATOMS * FDIM) {
        int n = idx >> 7, f = idx & 127;
        s[idx] = embed[(size_t)z[n] * FDIM + f];
        ds[idx] = 0.0f;
    }
}

// ---------------------------------------------------------------------------
// Message phase: w_s = (rbf@dw + db)*env; inv = phi[dst]*w_s; scatter to src
// ---------------------------------------------------------------------------
__global__ void edge_message(const float* __restrict__ rbfb, const float* __restrict__ envb,
                             const float* __restrict__ unitb, const float* __restrict__ phi,
                             const float* __restrict__ v, const int* __restrict__ nbrs,
                             const float* __restrict__ dw, const float* __restrict__ db,
                             float* __restrict__ ds_acc, float* __restrict__ dv_acc)
{
    int e = blockIdx.x, f = threadIdx.x;
    __shared__ float sr[NRBF];
    __shared__ float sgeo[4];   // env, unit xyz
    __shared__ int sidx[2];
    if (f < NRBF) sr[f] = rbfb[(size_t)e * NRBF + f];
    else if (f == NRBF) sgeo[0] = envb[e];
    else if (f < NRBF + 4) sgeo[f - NRBF] = unitb[(size_t)e * 3 + (f - NRBF - 1)];
    else if (f == NRBF + 4) sidx[0] = nbrs[2 * e];
    else if (f == NRBF + 5) sidx[1] = nbrs[2 * e + 1];
    __syncthreads();
    int src = sidx[0], dst = sidx[1];
    float w0 = db[f], w1 = db[FDIM + f], w2 = db[2 * FDIM + f];
    #pragma unroll
    for (int j = 0; j < NRBF; j++) {
        float rj = sr[j];
        const float* dwj = dw + (size_t)j * F3;
        w0 += rj * dwj[f]; w1 += rj * dwj[FDIM + f]; w2 += rj * dwj[2 * FDIM + f];
    }
    float ev = sgeo[0];
    w0 *= ev; w1 *= ev; w2 *= ev;
    const float* ph = phi + (size_t)dst * F3;
    float i0 = ph[f] * w0, i1 = ph[FDIM + f] * w1, i2 = ph[2 * FDIM + f] * w2;
    atomicAdd(&ds_acc[(size_t)src * FDIM + f], i1);
    const float* vd = v + (size_t)dst * F3;
    #pragma unroll
    for (int d = 0; d < 3; d++) {
        float dval = i2 * sgeo[1 + d] + i0 * vd[d * FDIM + f];
        atomicAdd(&dv_acc[(size_t)src * F3 + d * FDIM + f], dval);
    }
}

// s += ds; v += dv; re-zero accumulators; emit v_bf16 for the vector GEMMs
__global__ void apply_deltas(float* __restrict__ s, float* __restrict__ v,
                             float* __restrict__ ds, float* __restrict__ dv,
                             bf16_t* __restrict__ v_bf)
{
    int idx = blockIdx.x * blockDim.x + threadIdx.x;
    if (idx >= N_ATOMS * F3) return;
    float nv = v[idx] + dv[idx];
    v[idx] = nv; dv[idx] = 0.0f; v_bf[idx] = f2bf(nv);
    if (idx < N_ATOMS * FDIM) { s[idx] += ds[idx]; ds[idx] = 0.0f; }
}

// stack = concat(s, vnorm(v_v))  (bf16, N x 256)
__global__ void stack_build(const float* __restrict__ s, const float* __restrict__ vv,
                            bf16_t* __restrict__ stack_bf)
{
    int idx = blockIdx.x * blockDim.x + threadIdx.x;
    if (idx >= N_ATOMS * FDIM) return;
    int n = idx >> 7, g = idx & 127;
    float sum = 3.0f * 1e-15f;
    #pragma unroll
    for (int d = 0; d < 3; d++) { float t = vv[(size_t)n * F3 + d * FDIM + g]; sum += t * t; }
    stack_bf[(size_t)n * 256 + g]       = f2bf(s[idx]);
    stack_bf[(size_t)n * 256 + 128 + g] = f2bf(sqrtf(sum));
}

// v += u_v*a_vv ; s += dot(u_v,v_v)*a_sv + a_ss
__global__ void update_combine(float* __restrict__ s, float* __restrict__ v,
                               const float* __restrict__ uv, const float* __restrict__ vv,
                               const float* __restrict__ split)
{
    int idx = blockIdx.x * blockDim.x + threadIdx.x;
    if (idx >= N_ATOMS * FDIM) return;
    int n = idx >> 7, f = idx & 127;
    const float* sp = split + (size_t)n * F3;
    float avv = sp[f], asv = sp[FDIM + f], ass = sp[2 * FDIM + f];
    float dot = 0.0f;
    #pragma unroll
    for (int d = 0; d < 3; d++) {
        size_t o = (size_t)n * F3 + d * FDIM + f;
        float u = uv[o];
        dot += u * vv[o];
        v[o] += u * avv;
    }
    s[idx] += dot * asv + ass;
}

// pair = s[src] + s[dst] (bf16)
__global__ void build_pair(const float* __restrict__ s, const int* __restrict__ nbrs,
                           bf16_t* __restrict__ pair_bf)
{
    int idx = blockIdx.x * blockDim.x + threadIdx.x;
    if (idx >= N_EDGES * FDIM) return;
    int e = idx >> 7, f = idx & 127;
    int src = nbrs[2 * e], dst = nbrs[2 * e + 1];
    pair_bf[idx] = f2bf(s[(size_t)src * FDIM + f] + s[(size_t)dst * FDIM + f]);
}

__global__ void zero_out_k(float* __restrict__ out)
{
    int idx = blockIdx.x * blockDim.x + threadIdx.x;
    if (idx < N_ATOMS * 3) out[idx] = 0.0f;
}

// f_edge = (h @ ro_w2 + ro_b2) * unit ; out[src]+=f ; out[dst]-=f
__global__ void readout_scatter(const bf16_t* __restrict__ h, const float* __restrict__ ro_w2,
                                const float* __restrict__ ro_b2, const float* __restrict__ unitb,
                                const int* __restrict__ nbrs, float* __restrict__ out)
{
    int e = blockIdx.x * blockDim.x + threadIdx.x;
    if (e >= N_EDGES) return;
    const bf16_t* hr = h + (size_t)e * 256;
    float acc = ro_b2[0];
    #pragma unroll 8
    for (int k = 0; k < 256; k++) acc += bf2f(hr[k]) * ro_w2[k];
    int src = nbrs[2 * e], dst = nbrs[2 * e + 1];
    #pragma unroll
    for (int d = 0; d < 3; d++) {
        float fd = acc * unitb[(size_t)e * 3 + d];
        atomicAdd(&out[(size_t)src * 3 + d], fd);
        atomicAdd(&out[(size_t)dst * 3 + d], -fd);
    }
}

// ---------------------------------------------------------------------------
extern "C" void kernel_launch(void* const* d_in, const int* in_sizes, int n_in,
                              void* d_out, int out_size, void* d_ws, size_t ws_size,
                              hipStream_t stream)
{
    (void)in_sizes; (void)n_in; (void)out_size; (void)ws_size;
    const float* xyz      = (const float*)d_in[0];
    const int*   z        = (const int*)d_in[1];
    const int*   nbrs     = (const int*)d_in[2];
    const float* embed    = (const float*)d_in[3];
    const float* de_w     = (const float*)d_in[4];
    const float* de_b     = (const float*)d_in[5];
    const float* msg_w1   = (const float*)d_in[6];
    const float* msg_b1   = (const float*)d_in[7];
    const float* msg_w2   = (const float*)d_in[8];
    const float* msg_b2   = (const float*)d_in[9];
    const float* msg_dw   = (const float*)d_in[10];
    const float* msg_db   = (const float*)d_in[11];
    const float* upd_u    = (const float*)d_in[12];
    const float* upd_v    = (const float*)d_in[13];
    const float* upd_w1   = (const float*)d_in[14];
    const float* upd_b1   = (const float*)d_in[15];
    const float* upd_w2   = (const float*)d_in[16];
    const float* upd_b2   = (const float*)d_in[17];
    const float* edge_w1  = (const float*)d_in[18];
    const float* edge_b1  = (const float*)d_in[19];
    const float* edge_w2  = (const float*)d_in[20];
    const float* edge_b2  = (const float*)d_in[21];
    const float* ro_w1    = (const float*)d_in[22];
    const float* ro_b1    = (const float*)d_in[23];
    const float* ro_w2    = (const float*)d_in[24];
    const float* ro_b2    = (const float*)d_in[25];
    float* out = (float*)d_out;

    // ---- workspace layout ----
    size_t off = 0;
    char* base = (char*)d_ws;
    auto alloc = [&](size_t bytes) -> char* {
        char* p = base + off; off += (bytes + 255) & ~(size_t)255; return p;
    };
    float*  unitb  = (float*)alloc((size_t)N_EDGES * 3 * 4);
    float*  rbfb   = (float*)alloc((size_t)N_EDGES * NRBF * 4);
    float*  envb   = (float*)alloc((size_t)N_EDGES * 4);
    float*  eij    = (float*)alloc((size_t)N_EDGES * FDIM * 4);
    float*  s      = (float*)alloc((size_t)N_ATOMS * FDIM * 4);
    float*  v      = (float*)alloc((size_t)N_ATOMS * F3 * 4);
    float*  ds     = (float*)alloc((size_t)N_ATOMS * FDIM * 4);
    float*  dv     = (float*)alloc((size_t)N_ATOMS * F3 * 4);
    float*  phi    = (float*)alloc((size_t)N_ATOMS * F3 * 4);
    float*  uvb    = (float*)alloc((size_t)N_ATOMS * F3 * 4);
    float*  vvb    = (float*)alloc((size_t)N_ATOMS * F3 * 4);
    float*  usplit = (float*)alloc((size_t)N_ATOMS * F3 * 4);
    bf16_t* s_bf   = (bf16_t*)alloc((size_t)N_ATOMS * FDIM * 2);
    bf16_t* h_bf   = (bf16_t*)alloc((size_t)N_ATOMS * FDIM * 2);
    bf16_t* v_bf   = (bf16_t*)alloc((size_t)N_ATOMS * F3 * 2);
    bf16_t* stk_bf = (bf16_t*)alloc((size_t)N_ATOMS * 256 * 2);
    bf16_t* pair_bf= (bf16_t*)alloc((size_t)N_EDGES * FDIM * 2);   // reused as e_bf
    bf16_t* ehid_bf= (bf16_t*)alloc((size_t)N_EDGES * 256 * 2);    // reused by readout
    bf16_t* msg_w1T = (bf16_t*)alloc((size_t)NLAY * FDIM * FDIM * 2);
    bf16_t* msg_w2T = (bf16_t*)alloc((size_t)NLAY * F3 * FDIM * 2);
    bf16_t* upd_uT  = (bf16_t*)alloc((size_t)NLAY * FDIM * FDIM * 2);
    bf16_t* upd_vT  = (bf16_t*)alloc((size_t)NLAY * FDIM * FDIM * 2);
    bf16_t* upd_w1T = (bf16_t*)alloc((size_t)NLAY * FDIM * 256 * 2);
    bf16_t* upd_w2T = (bf16_t*)alloc((size_t)NLAY * F3 * FDIM * 2);
    bf16_t* edge_w1T= (bf16_t*)alloc((size_t)NLAY * 256 * FDIM * 2);
    bf16_t* edge_w2T= (bf16_t*)alloc((size_t)NLAY * FDIM * 256 * 2);
    bf16_t* ro_w1T  = (bf16_t*)alloc((size_t)256 * FDIM * 2);

    // ---- weight transposes (f32 -> bf16 [Nc x K]) ----
    auto wtr = [&](const float* W, bf16_t* WT, int K, int Nc) {
        w_transpose<<<CDIV(K * Nc, 256), 256, 0, stream>>>(W, WT, K, Nc);
    };
    for (int l = 0; l < NLAY; l++) {
        wtr(msg_w1  + (size_t)l * FDIM * FDIM, msg_w1T  + (size_t)l * FDIM * FDIM, FDIM, FDIM);
        wtr(msg_w2  + (size_t)l * FDIM * F3,   msg_w2T  + (size_t)l * F3 * FDIM,   FDIM, F3);
        wtr(upd_u   + (size_t)l * FDIM * FDIM, upd_uT   + (size_t)l * FDIM * FDIM, FDIM, FDIM);
        wtr(upd_v   + (size_t)l * FDIM * FDIM, upd_vT   + (size_t)l * FDIM * FDIM, FDIM, FDIM);
        wtr(upd_w1  + (size_t)l * 256 * FDIM,  upd_w1T  + (size_t)l * FDIM * 256,  256, FDIM);
        wtr(upd_w2  + (size_t)l * FDIM * F3,   upd_w2T  + (size_t)l * F3 * FDIM,   FDIM, F3);
        wtr(edge_w1 + (size_t)l * FDIM * 256,  edge_w1T + (size_t)l * 256 * FDIM,  FDIM, 256);
        wtr(edge_w2 + (size_t)l * 256 * FDIM,  edge_w2T + (size_t)l * FDIM * 256,  256, FDIM);
    }
    wtr(ro_w1, ro_w1T, FDIM, 256);

    // ---- geometry, e_ij init, node init ----
    edge_geom<<<CDIV(N_EDGES, 256), 256, 0, stream>>>(xyz, nbrs, unitb, rbfb, envb);
    e_init<<<N_EDGES, FDIM, 0, stream>>>(rbfb, envb, de_w, de_b, eij);
    init_nodes<<<CDIV(N_ATOMS * F3, 256), 256, 0, stream>>>(embed, z, s, v, ds, dv);

    // waves = (M/32) * (Nc/64); 8 waves (256 thr) per block
    auto gemm_blocks = [](int M, int Nc) { return CDIV((M / 32) * (Nc / 64), 8); };

    for (int l = 0; l < NLAY; l++) {
        const bf16_t* mw1 = msg_w1T  + (size_t)l * FDIM * FDIM;
        const bf16_t* mw2 = msg_w2T  + (size_t)l * F3 * FDIM;
        const bf16_t* uu  = upd_uT   + (size_t)l * FDIM * FDIM;
        const bf16_t* uv_ = upd_vT   + (size_t)l * FDIM * FDIM;
        const bf16_t* uw1 = upd_w1T  + (size_t)l * FDIM * 256;
        const bf16_t* uw2 = upd_w2T  + (size_t)l * F3 * FDIM;
        const bf16_t* ew1 = edge_w1T + (size_t)l * 256 * FDIM;
        const bf16_t* ew2 = edge_w2T + (size_t)l * FDIM * 256;

        // phi = swish(s @ msg_w1 + b1) @ msg_w2 + b2
        f32_to_bf16<<<CDIV(N_ATOMS * FDIM, 256), 256, 0, stream>>>(s, s_bf, N_ATOMS * FDIM);
        wmma_gemm<128, true, true, false><<<gemm_blocks(N_ATOMS, 128), 256, 0, stream>>>(
            s_bf, FDIM, mw1, msg_b1 + (size_t)l * FDIM, nullptr, h_bf, FDIM, N_ATOMS, FDIM);
        wmma_gemm<128, false, false, false><<<gemm_blocks(N_ATOMS, F3), 256, 0, stream>>>(
            h_bf, FDIM, mw2, msg_b2 + (size_t)l * F3, phi, nullptr, F3, N_ATOMS, F3);

        // per-edge message + scatter
        edge_message<<<N_EDGES, FDIM, 0, stream>>>(rbfb, envb, unitb, phi, v, nbrs,
            msg_dw + (size_t)l * NRBF * F3, msg_db + (size_t)l * F3, ds, dv);
        apply_deltas<<<CDIV(N_ATOMS * F3, 256), 256, 0, stream>>>(s, v, ds, dv, v_bf);

        // u_v / v_v: (3N x 128) @ (128 x 128)
        wmma_gemm<128, false, false, false><<<gemm_blocks(3 * N_ATOMS, 128), 256, 0, stream>>>(
            v_bf, FDIM, uu,  nullptr, uvb, nullptr, FDIM, 3 * N_ATOMS, FDIM);
        wmma_gemm<128, false, false, false><<<gemm_blocks(3 * N_ATOMS, 128), 256, 0, stream>>>(
            v_bf, FDIM, uv_, nullptr, vvb, nullptr, FDIM, 3 * N_ATOMS, FDIM);

        // update MLP
        stack_build<<<CDIV(N_ATOMS * FDIM, 256), 256, 0, stream>>>(s, vvb, stk_bf);
        wmma_gemm<256, true, true, false><<<gemm_blocks(N_ATOMS, 128), 256, 0, stream>>>(
            stk_bf, 256, uw1, upd_b1 + (size_t)l * FDIM, nullptr, h_bf, FDIM, N_ATOMS, FDIM);
        wmma_gemm<128, false, false, false><<<gemm_blocks(N_ATOMS, F3), 256, 0, stream>>>(
            h_bf, FDIM, uw2, upd_b2 + (size_t)l * F3, usplit, nullptr, F3, N_ATOMS, F3);
        update_combine<<<CDIV(N_ATOMS * FDIM, 256), 256, 0, stream>>>(s, v, uvb, vvb, usplit);

        // edge MLP (the big GEMMs): e_ij += swish(pair@W1+b1)@W2+b2
        build_pair<<<CDIV(N_EDGES * FDIM, 256), 256, 0, stream>>>(s, nbrs, pair_bf);
        wmma_gemm<128, true, true, false><<<gemm_blocks(N_EDGES, 256), 256, 0, stream>>>(
            pair_bf, FDIM, ew1, edge_b1 + (size_t)l * 256, nullptr, ehid_bf, 256, N_EDGES, 256);
        wmma_gemm<256, false, false, true><<<gemm_blocks(N_EDGES, 128), 256, 0, stream>>>(
            ehid_bf, 256, ew2, edge_b2 + (size_t)l * FDIM, eij, nullptr, FDIM, N_EDGES, FDIM);
    }

    // readout: h = swish(e_ij @ ro_w1 + b1); f_edge = (h@ro_w2+b2)*unit; scatter
    f32_to_bf16<<<CDIV(N_EDGES * FDIM, 256), 256, 0, stream>>>(eij, pair_bf, N_EDGES * FDIM);
    wmma_gemm<128, true, true, false><<<gemm_blocks(N_EDGES, 256), 256, 0, stream>>>(
        pair_bf, FDIM, ro_w1T, ro_b1, nullptr, ehid_bf, 256, N_EDGES, 256);
    zero_out_k<<<CDIV(N_ATOMS * 3, 256), 256, 0, stream>>>(out);
    readout_scatter<<<CDIV(N_EDGES, 256), 256, 0, stream>>>(ehid_bf, ro_w2, ro_b2,
                                                            unitb, nbrs, out);
}
